// Network_clip_dgcnn_acc_24713241821965
// MI455X (gfx1250) — matrix operand order repaired and checked by reading, not compiled, not run on hardware
//
#include <hip/hip_runtime.h>
#include <hip/hip_bf16.h>
#include <cstdint>

typedef __attribute__((ext_vector_type(16))) _Float16     v16h;
typedef __attribute__((ext_vector_type(8)))  float        v8f;
typedef __attribute__((ext_vector_type(8)))  unsigned int v8u;
typedef __attribute__((ext_vector_type(4)))  unsigned int v4u;
typedef __attribute__((ext_vector_type(8)))  int          v8i;
typedef __attribute__((ext_vector_type(4)))  int          v4i;

#define NN   512
#define DD   320
#define EPS  1e-5f
#define SLP  0.01f

// workspace layout (bytes)
#define UA_OFF  0
#define UB_OFF  (NN * DD * 4)          // 655360
#define W2T_OFF (2 * NN * DD * 4)      // 1310720 ; W2T is D x D f16, transposed: W2T[n][k]

#define TILE_BYTES 10240                              // 16 cols x 320 K x f16
#define SMEM_BYTES (2 * TILE_BYTES + 8 * (DD * 16 * 4))  // 2 B-tile bufs + 8 waves * 20KB = 184320

#if __has_builtin(__builtin_amdgcn_tensor_load_to_lds) && __has_builtin(__builtin_amdgcn_s_wait_tensorcnt)
#define HAVE_TDM 1
#else
#define HAVE_TDM 0
#endif

// ---------------------------------------------------------------------------
// Prep: UA[i,n] = Phi_A[i,:] @ W1[0:320, n];  UB[j,n] = Phi_B[j,:] @ W1[320:640, n]
// ---------------------------------------------------------------------------
__global__ void prep_uab_kernel(const float* __restrict__ robot,
                                const float* __restrict__ object,
                                const float* __restrict__ lang,
                                const float* __restrict__ W1,
                                float* __restrict__ UA, float* __restrict__ UB) {
  int idx   = blockIdx.x * blockDim.x + threadIdx.x;   // 2*512*320 threads exactly
  int which = idx / (NN * DD);
  int rem   = idx - which * (NN * DD);
  int i     = rem / DD;
  int n     = rem - i * DD;
  const float* emb  = which ? object : robot;
  const float* wcol = W1 + (size_t)which * DD * DD + n;   // W1[(which*320 + d)*320 + n]
  float acc = 0.f;
#pragma unroll 4
  for (int d = 0; d < DD; ++d) {
    float phi = (d < 256) ? emb[i * 256 + d] : lang[d - 256];
    acc += phi * wcol[(size_t)d * DD];
  }
  (which ? UB : UA)[i * DD + n] = acc;
}

// W2T[n*320 + k] = (f16) W2[k*320 + n]  -> each 16-col tile is a contiguous 10KB block
__global__ void conv_w2t_kernel(const float* __restrict__ W2, _Float16* __restrict__ W2t) {
  int idx = blockIdx.x * blockDim.x + threadIdx.x;     // 320*320 threads exactly
  int n = idx / DD;
  int k = idx - n * DD;
  W2t[idx] = (_Float16)W2[k * DD + n];
}

// ---------------------------------------------------------------------------
// TDM: 1-D copy of one contiguous 10KB W2T tile into LDS (D# per ISA ch.8).
// data_size=4B, tensor_dim0=tile_dim0=2560 dwords, 1 row.
// ---------------------------------------------------------------------------
__device__ __forceinline__ void tdm_load_tile(const _Float16* gsrc, unsigned lds_off) {
#if HAVE_TDM
  unsigned long long ga = (unsigned long long)(uintptr_t)gsrc;
  v4u g0 = { 1u,                                   // count=1, is_restore=0, gather=0
             lds_off,                              // lds_addr (bytes)
             (unsigned)ga,                         // global_addr[31:0]
             (unsigned)((ga >> 32) & 0x1FFFFFFu) | (2u << 30) };  // ga[56:32] | type=2
  v8i g1 = { (int)(2u << 16),                      // wg_mask=0, data_size=2 (4B)
             (int)(2560u << 16),                   // tensor_dim0[15:0] << 16 (bits 79:48)
             (int)(1u << 16),                      // tensor_dim0 hi=0 | tensor_dim1=1
             (int)(2560u << 16),                   // tensor_dim1 hi=0 | tile_dim0=2560
             1,                                    // tile_dim1=1, tile_dim2=0
             2560,                                 // tensor_dim0_stride[31:0]
             (int)(2560u << 16),                   // stride0 hi=0 | tensor_dim1_stride lo
             0 };
  v4i z4 = { 0, 0, 0, 0 };
  v8i z8 = { 0, 0, 0, 0, 0, 0, 0, 0 };
  __builtin_amdgcn_tensor_load_to_lds(g0, g1, z4, z4, z8, 0);  // 6-arg (clang-23) form
#else
  (void)gsrc; (void)lds_off;
#endif
}

// 16 contiguous f16 -> two aligned ds_load_b128
__device__ __forceinline__ v8u load_frag16u(const _Float16* p) {
  const v4u* q = (const v4u*)p;
  v4u lo = q[0], hi = q[1];
  return __builtin_shufflevector(lo, hi, 0, 1, 2, 3, 4, 5, 6, 7);
}

// ---------------------------------------------------------------------------
// Main fused kernel: per wave, 16 pairs (fixed i, 16 consecutive j).
// __launch_bounds__(256, 1): LDS already caps occupancy at 1 WG/WGP, so let the
// allocator take ~200 VGPRs (80 A-frags + 80 B-frags + acc) without spilling.
// ---------------------------------------------------------------------------
__global__ void __launch_bounds__(256, 1)
pair_mlp_kernel(const float* __restrict__ UA, const float* __restrict__ UB,
                const _Float16* __restrict__ W2t,
                const float* __restrict__ b1, const float* __restrict__ g1,
                const float* __restrict__ be1,
                const float* __restrict__ b2, const float* __restrict__ g2,
                const float* __restrict__ be2,
                const float* __restrict__ W3, const float* __restrict__ b3,
                float* __restrict__ out) {
  extern __shared__ char smem[];
  const int tid  = threadIdx.x;
  const int wave = tid >> 5;
  const int lane = tid & 31;
  const int half = lane >> 4;
  const int r    = lane & 15;

  float*    myh2 = (float*)(smem + 2 * TILE_BYTES + wave * (DD * 16 * 4));
  _Float16* myh1 = (_Float16*)myh2;                 // overlapped: h1 dead before h2 written

  const int t     = blockIdx.x * 8 + wave;          // pair-tile id, 0..16383
  const int i     = t >> 5;
  const int jbase = (t & 31) << 4;

  const float* ua = UA + (size_t)i * DD;

  // ---- Stage 1: h1 rows (LN1 + leaky) -> f16 in LDS ----
  for (int m = 0; m < 16; ++m) {
    const float* ub = UB + (size_t)(jbase + m) * DD;
    float xv[10];
    float s = 0.f, s2 = 0.f;
#pragma unroll
    for (int q = 0; q < 10; ++q) {
      int c   = lane + 32 * q;
      float x = ua[c] + ub[c] + b1[c];
      xv[q] = x; s += x; s2 += x * x;
    }
#pragma unroll
    for (int off = 16; off > 0; off >>= 1) {
      s  += __shfl_xor(s,  off, 32);
      s2 += __shfl_xor(s2, off, 32);
    }
    float mu = s * (1.f / 320.f);
    float rs = rsqrtf(s2 * (1.f / 320.f) - mu * mu + EPS);
#pragma unroll
    for (int q = 0; q < 10; ++q) {
      int c   = lane + 32 * q;
      float v = (xv[q] - mu) * rs * g1[c] + be1[c];
      v = v > 0.f ? v : SLP * v;
      myh1[m * DD + c] = (_Float16)v;
    }
  }
  __syncthreads();

  // ---- Stage 2: A fragments (contiguous k = 16*half + s order) into registers ----
  v16h afrag[10];
#pragma unroll
  for (int kc = 0; kc < 10; ++kc) {
    const v4u* lo = (const v4u*)(myh1 + r * DD + kc * 32 + half * 16);
    const v4u* hi = (const v4u*)(myh1 + r * DD + kc * 32 + half * 16 + 8);
    afrag[kc] = __builtin_bit_cast(v16h,
        __builtin_shufflevector(lo[0], hi[0], 0, 1, 2, 3, 4, 5, 6, 7));
  }

  const unsigned smem_base = (unsigned)(uintptr_t)(void*)smem;  // low 32b = LDS offset

  // ---- Stage 3: GEMM over 20 N-tiles, TDM double-buffered B tiles ----
#if HAVE_TDM
  if (wave == 0) tdm_load_tile(W2t, smem_base);       // T(0) -> buf0
#endif
  for (int nt = 0; nt < 20; ++nt) {
#if HAVE_TDM
    if (wave == 0) __builtin_amdgcn_s_wait_tensorcnt(0);   // T(nt) landed
    __syncthreads();                                       // buf[nt&1] visible; prev reads done
    if (wave == 0 && nt < 19)                              // overlap T(nt+1) with compute(nt)
      tdm_load_tile(W2t + (size_t)(nt + 1) * (TILE_BYTES / 2),
                    smem_base + ((nt + 1) & 1) * TILE_BYTES);
    const _Float16* bt = (const _Float16*)(smem + (nt & 1) * TILE_BYTES);
#else
    __syncthreads();
    {   // two-phase cooperative copy: batch global loads, then LDS stores
      unsigned int tmp[10];
      const unsigned int* gs = (const unsigned int*)(W2t + (size_t)nt * (TILE_BYTES / 2));
#pragma unroll
      for (int it = 0; it < 10; ++it) tmp[it] = gs[tid + 256 * it];
      unsigned int* lb = (unsigned int*)smem;
#pragma unroll
      for (int it = 0; it < 10; ++it) lb[tid + 256 * it] = tmp[it];
    }
    __syncthreads();
    const _Float16* bt = (const _Float16*)smem;
#endif
    // Batch-preload ALL 10 B fragments (20 ds_load_b128 back-to-back), pinned by one
    // asm barrier: one s_wait_dscnt, then 10 uninterrupted v_wmma ops.
    v8u bfr[10];
#pragma unroll
    for (int kc = 0; kc < 10; ++kc)
      bfr[kc] = load_frag16u(bt + r * DD + kc * 32 + half * 16);  // 16 contiguous f16

    asm volatile("" : "+v"(bfr[0]), "+v"(bfr[1]), "+v"(bfr[2]), "+v"(bfr[3]),
                      "+v"(bfr[4]), "+v"(bfr[5]), "+v"(bfr[6]), "+v"(bfr[7]),
                      "+v"(bfr[8]), "+v"(bfr[9]));

    v8f acc = {};
#pragma unroll
    for (int kc = 0; kc < 10; ++kc)
      acc = __builtin_amdgcn_wmma_f32_16x16x32_f16(
          /*neg_a=*/false, afrag[kc], /*neg_b=*/false, __builtin_bit_cast(v16h, bfr[kc]),
          /*c_mod=*/(short)0, acc, /*reuse_a=*/false, /*reuse_b=*/false);

    // D layout: VGPR d -> row (d + 8*half), col r; fold in b2
    int   ncol  = nt * 16 + r;
    float bias2 = b2[ncol];
#pragma unroll
    for (int d = 0; d < 8; ++d)
      myh2[(d + 8 * half) * DD + ncol] = acc[d] + bias2;
  }

  // ---- Stage 4: LN2 + leaky + dot(W3) + |.| ----
  for (int m = 0; m < 16; ++m) {
    float hv[10];
    float s = 0.f, s2 = 0.f;
#pragma unroll
    for (int q = 0; q < 10; ++q) {
      int c   = lane + 32 * q;
      float x = myh2[m * DD + c];
      hv[q] = x; s += x; s2 += x * x;
    }
#pragma unroll
    for (int off = 16; off > 0; off >>= 1) {
      s  += __shfl_xor(s,  off, 32);
      s2 += __shfl_xor(s2, off, 32);
    }
    float mu = s * (1.f / 320.f);
    float rs = rsqrtf(s2 * (1.f / 320.f) - mu * mu + EPS);
    float dot = 0.f;
#pragma unroll
    for (int q = 0; q < 10; ++q) {
      int c   = lane + 32 * q;
      float v = (hv[q] - mu) * rs * g2[c] + be2[c];
      v = v > 0.f ? v : SLP * v;
      dot += v * W3[c];
    }
#pragma unroll
    for (int off = 16; off > 0; off >>= 1) dot += __shfl_xor(dot, off, 32);
    if (lane == 0) out[(size_t)t * 16 + m] = fabsf(dot + b3[0]);  // t*16+m == i*512+j
  }
}

// ---------------------------------------------------------------------------
extern "C" void kernel_launch(void* const* d_in, const int* in_sizes, int n_in,
                              void* d_out, int out_size, void* d_ws, size_t ws_size,
                              hipStream_t stream) {
  const float* robot  = (const float*)d_in[0];
  const float* object = (const float*)d_in[1];
  const float* lang   = (const float*)d_in[2];
  const float* W1     = (const float*)d_in[3];
  const float* b1     = (const float*)d_in[4];
  const float* g1     = (const float*)d_in[5];
  const float* be1    = (const float*)d_in[6];
  const float* W2     = (const float*)d_in[7];
  const float* b2     = (const float*)d_in[8];
  const float* g2     = (const float*)d_in[9];
  const float* be2    = (const float*)d_in[10];
  const float* W3     = (const float*)d_in[11];
  const float* b3     = (const float*)d_in[12];
  float* out = (float*)d_out;

  char* ws = (char*)d_ws;
  float*    UA  = (float*)(ws + UA_OFF);
  float*    UB  = (float*)(ws + UB_OFF);
  _Float16* W2t = (_Float16*)(ws + W2T_OFF);

  prep_uab_kernel<<<(2 * NN * DD) / 256, 256, 0, stream>>>(robot, object, lang, W1, UA, UB);
  conv_w2t_kernel<<<(DD * DD) / 256, 256, 0, stream>>>(W2, W2t);

  (void)hipFuncSetAttribute((const void*)pair_mlp_kernel,
                            hipFuncAttributeMaxDynamicSharedMemorySize, SMEM_BYTES);
  pair_mlp_kernel<<<16384 / 8, 256, SMEM_BYTES, stream>>>(
      UA, UB, W2t, b1, g1, be1, b2, g2, be2, W3, b3, out);
}
